// MultiscaleRGCN_56066503082342
// MI455X (gfx1250) — matrix operand order repaired and checked by reading, not compiled, not run on hardware
//
#include <hip/hip_runtime.h>
#include <hip/hip_bf16.h>

// ---------------------------------------------------------------------------
// MultiscaleRGCN for MI455X (gfx1250, wave32).
// GEMMs run on v_wmma_f32_16x16x32_bf16 (fp32 data, bf16 fragments, f32 acc).
// Edge aggregation = fp32 atomic scatter-add (L2-resident agg buffer).
// ---------------------------------------------------------------------------

typedef __attribute__((ext_vector_type(16))) __bf16 v16bf;
typedef __attribute__((ext_vector_type(8)))  __bf16 bf16x8;
typedef __attribute__((ext_vector_type(4)))  __bf16 bf16x4;
typedef __attribute__((ext_vector_type(8)))  float  v8f;

static constexpr int N_   = 20000;
static constexpr int E_   = 320000;
static constexpr int R_   = 8;
static constexpr int L_   = 3;
static constexpr int IN_  = 128;
static constexpr int H_   = 256;
static constexpr int A1_  = H_ / 4;   // 64
static constexpr int OUT_ = 128;

// ------------------------------ utility ------------------------------------

__global__ void k_zero4(float4* __restrict__ p, int n4) {
  int i = blockIdx.x * blockDim.x + threadIdx.x;
  if (i < n4) p[i] = make_float4(0.f, 0.f, 0.f, 0.f);
}

// counts[dst*R + etype] += 1
__global__ void k_count(const int* __restrict__ dst, const int* __restrict__ et,
                        float* __restrict__ counts) {
  int e = blockIdx.x * blockDim.x + threadIdx.x;
  if (e < E_) atomicAdd(&counts[dst[e] * R_ + et[e]], 1.0f);
}

// agg[(dst*R + etype), :] += h[src, :]   (one wave32 per edge, 8 feats/lane)
__global__ void k_scatter(const float* __restrict__ h, const int* __restrict__ src,
                          const int* __restrict__ dst, const int* __restrict__ et,
                          float* __restrict__ agg) {
  int e    = blockIdx.x * (blockDim.x >> 5) + (threadIdx.x >> 5);
  int lane = threadIdx.x & 31;
  if (e >= E_) return;
  int s = src[e], d = dst[e], r = et[e];
  const float* hs = h + (size_t)s * H_;
  float* ag = agg + ((size_t)d * R_ + r) * H_;
#pragma unroll
  for (int j = lane; j < H_; j += 32) atomicAdd(&ag[j], hs[j]);
}

// per-relation mean: agg[i] /= max(counts[i / H], 1)
__global__ void k_divide(float* __restrict__ agg, const float* __restrict__ counts) {
  int i = blockIdx.x * blockDim.x + threadIdx.x;
  if (i >= N_ * R_ * H_) return;
  int nr = i >> 8;  // / H_
  agg[i] *= 1.0f / fmaxf(counts[nr], 1.0f);
}

// ------------------------- WMMA bf16 GEMM ----------------------------------
// C[M,Nc] = A[M,K] @ B[K,Nc] (+ bias[col]) (+ C if accum) (relu optional)
// A,B,C fp32 row-major; staged to LDS as bf16 (B transposed); f32 WMMA acc.
// Block: 256 thr = 8 waves; macro-tile 128x64; wave tile 32x32 (2x2 of 16x16).
// Requires: Nc % 64 == 0, K % 64 == 0 (true for all call sites).

__global__ __launch_bounds__(256)
void k_gemm(const float* __restrict__ A, const float* __restrict__ B,
            const float* __restrict__ bias, float* __restrict__ C,
            int M, int Nc, int K, int accum, int relu) {
  constexpr int BM = 128, BN = 64, BK = 64;
  constexpr int BKp = 72;  // padded k-stride (144 B = 36 banks): conflict-free
  __shared__ __bf16 As[BM * BKp];   // row-major  [m][k]   18.4 KB
  __shared__ __bf16 Bs[BN * BKp];   // TRANSPOSED [n][k]    9.2 KB

  const int tid  = threadIdx.x;
  const int lane = tid & 31;
  const int wid  = tid >> 5;   // 0..7
  const int wm   = wid & 3;    // wave row (4 x 32 = 128)
  const int wn   = wid >> 2;   // wave col (2 x 32 = 64)
  const int m0   = blockIdx.x * BM;
  const int n0   = blockIdx.y * BN;
  const bool full = (m0 + BM) <= M;   // uniform: 156/157 blocks take fast path

  const int mlane = lane & 15;
  const int half  = lane >> 4;       // 0 or 1

  v8f acc[2][2] = {};

  for (int k0 = 0; k0 < K; k0 += BK) {
    __syncthreads();
    // --- stage A tile (128x64 fp32 -> bf16), float4 loads, b64 LDS stores ---
    if (full) {
#pragma unroll
      for (int j = tid; j < BM * (BK / 4); j += 256) {
        int r  = j >> 4;            // 0..127
        int c4 = (j & 15) * 4;      // 0,4,...,60
        float4 v = *(const float4*)&A[(size_t)(m0 + r) * K + k0 + c4];
        bf16x4 p = { (__bf16)v.x, (__bf16)v.y, (__bf16)v.z, (__bf16)v.w };
        *(bf16x4*)&As[r * BKp + c4] = p;
      }
    } else {
#pragma unroll
      for (int j = tid; j < BM * (BK / 4); j += 256) {
        int r  = j >> 4;
        int c4 = (j & 15) * 4;
        int gr = m0 + r;
        float4 v = make_float4(0.f, 0.f, 0.f, 0.f);
        if (gr < M) v = *(const float4*)&A[(size_t)gr * K + k0 + c4];
        bf16x4 p = { (__bf16)v.x, (__bf16)v.y, (__bf16)v.z, (__bf16)v.w };
        *(bf16x4*)&As[r * BKp + c4] = p;
      }
    }
    // --- stage B tile (64x64 fp32 -> bf16, transposed): coalesced dword loads
#pragma unroll
    for (int j = tid; j < BN * (BK / 4); j += 256) {
      int c  = j & 63;            // output column
      int kq = (j >> 6) * 4;      // k quad base
      const float* bp = &B[(size_t)(k0 + kq) * Nc + n0 + c];
      bf16x4 p = { (__bf16)bp[0], (__bf16)bp[(size_t)Nc],
                   (__bf16)bp[(size_t)2 * Nc], (__bf16)bp[(size_t)3 * Nc] };
      *(bf16x4*)&Bs[c * BKp + kq] = p;
    }
    __syncthreads();

#pragma unroll
    for (int ks = 0; ks < BK / 32; ++ks) {
      const int kb0 = ks * 32;
      // A fragments: 16-bit A 16x32 layout — lane half selects K 0..7/8..15
      // (+16..23/24..31); both chunks contiguous -> ds_load_b128 each.
      union { v16bf v; bf16x8 h[2]; } af[2];
      union { v16bf v; bf16x8 h[2]; } bf[2];
      const int kb = half * 8;
#pragma unroll
      for (int tm = 0; tm < 2; ++tm) {
        int mrow = wm * 32 + tm * 16 + mlane;
        af[tm].h[0] = *(const bf16x8*)&As[mrow * BKp + kb0 + kb];
        af[tm].h[1] = *(const bf16x8*)&As[mrow * BKp + kb0 + 16 + kb];
      }
      // B fragments: 32x16 bf16 — lane holds col N=lane&15, K krb..krb+7 and
      // 16+krb..16+krb+7 (matches A); contiguous in transposed LDS -> 2x ds_load_b128.
      const int krb = half * 8;
#pragma unroll
      for (int tn = 0; tn < 2; ++tn) {
        int colb = wn * 32 + tn * 16 + mlane;
        bf[tn].h[0] = *(const bf16x8*)&Bs[colb * BKp + kb0 + krb];
        bf[tn].h[1] = *(const bf16x8*)&Bs[colb * BKp + kb0 + 16 + krb];
      }
#pragma unroll
      for (int tm = 0; tm < 2; ++tm)
#pragma unroll
        for (int tn = 0; tn < 2; ++tn)
          acc[tm][tn] = __builtin_amdgcn_wmma_f32_16x16x32_bf16(
              false, af[tm].v, false, bf[tn].v, (short)0, acc[tm][tn],
              false, false);
    }
  }

  // epilogue: D 16x16 f32 layout — VGPR v: lanes 0-15 M=v, lanes 16-31 M=8+v
#pragma unroll
  for (int tm = 0; tm < 2; ++tm) {
#pragma unroll
    for (int tn = 0; tn < 2; ++tn) {
      int col = n0 + wn * 32 + tn * 16 + mlane;
#pragma unroll
      for (int v = 0; v < 8; ++v) {
        int row = m0 + wm * 32 + tm * 16 + half * 8 + v;
        if (full || row < M) {
          size_t idx = (size_t)row * Nc + col;
          float val = acc[tm][tn][v];
          if (bias)  val += bias[col];
          if (accum) val += C[idx];
          if (relu && val < 0.0f) val = 0.0f;
          C[idx] = val;
        }
      }
    }
  }
}

// ------------------------------ BatchNorm ----------------------------------

// column sums / sumsq via per-block partials + atomics (coalesced: thread=col)
__global__ void k_bn_stats(const float* __restrict__ t, float* __restrict__ sums,
                           float* __restrict__ sqs) {
  int c = threadIdx.x;  // H_ = 256 = blockDim.x
  float s = 0.0f, q = 0.0f;
  for (int r = blockIdx.x; r < N_; r += gridDim.x) {
    float v = t[(size_t)r * H_ + c];
    s += v;
    q += v * v;
  }
  atomicAdd(&sums[c], s);
  atomicAdd(&sqs[c], q);
}

// h = leaky_relu( (t - mu) * rsqrt(var+eps) * gamma + beta ), also into reps[l]
__global__ void k_bn_apply(const float* __restrict__ t, const float* __restrict__ sums,
                           const float* __restrict__ sqs, const float* __restrict__ gamma,
                           const float* __restrict__ beta, float* __restrict__ h,
                           float* __restrict__ rep) {
  int i = blockIdx.x * blockDim.x + threadIdx.x;
  if (i >= N_ * H_) return;
  int c = i & (H_ - 1);
  const float invN = 1.0f / (float)N_;
  float mu  = sums[c] * invN;
  float var = sqs[c] * invN - mu * mu;
  float v = (t[i] - mu) * rsqrtf(var + 1e-5f) * gamma[c] + beta[c];
  v = (v >= 0.0f) ? v : 0.1f * v;
  h[i]   = v;
  rep[i] = v;
}

// ------------------------- attention / output ------------------------------

// scores[i] = hidden[i,:] . W2 + b2   (hidden already relu'd by GEMM epilogue)
__global__ void k_scores(const float* __restrict__ hidden, const float* __restrict__ W2,
                         const float* __restrict__ b2, float* __restrict__ scores) {
  int i = blockIdx.x * blockDim.x + threadIdx.x;
  if (i >= L_ * N_) return;
  const float* hr = hidden + (size_t)i * A1_;
  float s = 0.0f;
#pragma unroll
  for (int j = 0; j < A1_; ++j) s += hr[j] * W2[j];
  scores[i] = s + b2[0];
}

// softmax over L scales per node; emb = sum_l w_l * reps_l
__global__ void k_combine(const float* __restrict__ reps, const float* __restrict__ scores,
                          float* __restrict__ emb) {
  int i = blockIdx.x * blockDim.x + threadIdx.x;
  if (i >= N_ * H_) return;
  int n = i >> 8;  // / H_
  float s0 = scores[n], s1 = scores[N_ + n], s2 = scores[2 * N_ + n];
  float m  = fmaxf(s0, fmaxf(s1, s2));
  float e0 = __expf(s0 - m), e1 = __expf(s1 - m), e2 = __expf(s2 - m);
  float inv = 1.0f / (e0 + e1 + e2);
  const size_t NH = (size_t)N_ * H_;
  emb[i] = (reps[i] * e0 + reps[NH + i] * e1 + reps[2 * NH + i] * e2) * inv;
}

// L2 row-normalize: out = t / max(||t||, 1e-12), one block (OUT_ thr) per row
__global__ void k_l2norm(const float* __restrict__ t, float* __restrict__ out) {
  __shared__ float red[OUT_];
  int row = blockIdx.x;
  int c   = threadIdx.x;
  float v = t[(size_t)row * OUT_ + c];
  red[c] = v * v;
  __syncthreads();
#pragma unroll
  for (int s = OUT_ / 2; s > 0; s >>= 1) {
    if (c < s) red[c] += red[c + s];
    __syncthreads();
  }
  float inv = 1.0f / fmaxf(sqrtf(red[0]), 1e-12f);
  out[(size_t)row * OUT_ + c] = v * inv;
}

// ------------------------------ launcher -----------------------------------

extern "C" void kernel_launch(void* const* d_in, const int* in_sizes, int n_in,
                              void* d_out, int out_size, void* d_ws, size_t ws_size,
                              hipStream_t stream) {
  const float* x         = (const float*)d_in[0];
  const int*   edge_idx  = (const int*)  d_in[1];   // [2, E]
  const int*   edge_type = (const int*)  d_in[2];
  const float* W_in      = (const float*)d_in[3];
  const float* b_in      = (const float*)d_in[4];
  const float* conv_W    = (const float*)d_in[5];   // [L, R, H, H]
  const float* conv_root = (const float*)d_in[6];   // [L, H, H]
  const float* conv_b    = (const float*)d_in[7];   // [L, H]
  const float* bn_gamma  = (const float*)d_in[8];
  const float* bn_beta   = (const float*)d_in[9];
  const float* attn_W1   = (const float*)d_in[10];  // [H, H/4]
  const float* attn_b1   = (const float*)d_in[11];
  const float* attn_W2   = (const float*)d_in[12];  // [H/4, 1]
  const float* attn_b2   = (const float*)d_in[13];
  const float* W_out     = (const float*)d_in[14];  // [H, OUT]
  const float* b_out     = (const float*)d_in[15];
  float*       out       = (float*)d_out;

  const int* src = edge_idx;        // row 0
  const int* dst = edge_idx + E_;   // row 1

  // workspace layout (floats)
  float* ws = (float*)d_ws;
  size_t off = 0;
  float* h       = ws + off; off += (size_t)N_ * H_;        // [N, H]
  float* reps    = ws + off; off += (size_t)L_ * N_ * H_;   // [L, N, H]
  float* agg     = ws + off; off += (size_t)N_ * R_ * H_;   // [N, R, H]
  float* counts  = ws + off; off += (size_t)N_ * R_;        // [N, R]
  float* tmp     = ws + off; off += (size_t)N_ * H_;        // [N, H]
  float* hidden  = ws + off; off += (size_t)L_ * N_ * A1_;  // [L*N, H/4]
  float* scores  = ws + off; off += (size_t)L_ * N_;        // [L*N]
  float* emb     = ws + off; off += (size_t)N_ * H_;        // [N, H]
  float* out_tmp = ws + off; off += (size_t)N_ * OUT_;      // [N, OUT]
  float* bn_sum  = ws + off; off += H_;
  float* bn_sq   = ws + off; off += H_;

  auto zero = [&](float* p, int n) {  // n must be a multiple of 4 (it is)
    int n4 = n >> 2;
    k_zero4<<<(n4 + 255) / 256, 256, 0, stream>>>((float4*)p, n4);
  };
  auto gemm = [&](const float* A, const float* B, const float* bias, float* C,
                  int M, int Nc, int K, int accum, int relu) {
    dim3 grid((M + 127) / 128, Nc / 64);
    k_gemm<<<grid, 256, 0, stream>>>(A, B, bias, C, M, Nc, K, accum, relu);
  };

  // per-(dst, relation) edge counts (fixed across layers)
  zero(counts, N_ * R_);
  k_count<<<(E_ + 255) / 256, 256, 0, stream>>>(dst, edge_type, counts);

  // h = x @ W_in + b_in
  gemm(x, W_in, b_in, h, N_, H_, IN_, 0, 0);

  for (int l = 0; l < L_; ++l) {
    // per-relation mean aggregation
    zero(agg, N_ * R_ * H_);
    k_scatter<<<(E_ + 7) / 8, 256, 0, stream>>>(h, src, dst, edge_type, agg);
    k_divide<<<(N_ * R_ * H_ + 255) / 256, 256, 0, stream>>>(agg, counts);

    // tmp = agg2d @ conv_W[l] + conv_b[l]  ;  tmp += h @ conv_root[l]
    gemm(agg, conv_W + (size_t)l * R_ * H_ * H_, conv_b + (size_t)l * H_,
         tmp, N_, H_, R_ * H_, 0, 0);
    gemm(h, conv_root + (size_t)l * H_ * H_, nullptr,
         tmp, N_, H_, H_, 1, 0);

    // BatchNorm (batch stats) + leaky_relu(0.1) -> h, reps[l]
    zero(bn_sum, H_);
    zero(bn_sq, H_);
    k_bn_stats<<<512, H_, 0, stream>>>(tmp, bn_sum, bn_sq);
    k_bn_apply<<<(N_ * H_ + 255) / 256, 256, 0, stream>>>(
        tmp, bn_sum, bn_sq, bn_gamma + (size_t)l * H_, bn_beta + (size_t)l * H_,
        h, reps + (size_t)l * N_ * H_);
  }

  // attention over scales
  gemm(reps, attn_W1, attn_b1, hidden, L_ * N_, A1_, H_, 0, 1);  // relu fused
  k_scores<<<(L_ * N_ + 255) / 256, 256, 0, stream>>>(hidden, attn_W2, attn_b2, scores);
  k_combine<<<(N_ * H_ + 255) / 256, 256, 0, stream>>>(reps, scores, emb);

  // output projection + row L2 normalize
  gemm(emb, W_out, b_out, out_tmp, N_, OUT_, H_, 0, 0);
  k_l2norm<<<N_, OUT_, 0, stream>>>(out_tmp, out);
}
